// MixLoraSparseMoe_10831907521052
// MI455X (gfx1250) — matrix operand order repaired and checked by reading, not compiled, not run on hardware
//
#include <hip/hip_runtime.h>

// ---------------------------------------------------------------------------
// Problem constants (match reference setup_inputs)
// ---------------------------------------------------------------------------
constexpr int BB   = 2;
constexpr int SS   = 1024;
constexpr int DD   = 1024;   // model dim
constexpr int HH   = 4096;   // hidden dim
constexpr int EE   = 8;      // experts
constexpr int RR   = 16;     // lora rank
constexpr int NTOK = BB * SS;        // 2048 tokens
constexpr int TEXP = NTOK * 2;       // 4096 expanded (token, k-slot) rows
constexpr float LSCALE = 32.0f / 16.0f;  // lora_alpha / lora_r

typedef __attribute__((ext_vector_type(16))) __bf16 v16bf;
typedef __attribute__((ext_vector_type(8)))  __bf16 v8bf;
typedef __attribute__((ext_vector_type(8)))  float  v8f;

// ---------------------------------------------------------------------------
// bf16 helpers (RNE)
// ---------------------------------------------------------------------------
__device__ __forceinline__ unsigned short f32_to_bf16(float f) {
    union { float f; unsigned int u; } v; v.f = f;
    unsigned int u = v.u;
    unsigned int r = u + 0x7FFFu + ((u >> 16) & 1u);
    return (unsigned short)(r >> 16);
}
__device__ __forceinline__ float bf16_to_f32(unsigned short h) {
    union { unsigned int u; float f; } v; v.u = ((unsigned int)h) << 16;
    return v.f;
}

// ---------------------------------------------------------------------------
// Elementwise f32 -> bf16 convert
// ---------------------------------------------------------------------------
__global__ void __launch_bounds__(256)
cvt_f32_bf16_kernel(const float* __restrict__ in, unsigned short* __restrict__ out, int n) {
    int i = blockIdx.x * 256 + threadIdx.x;
    if (i < n) out[i] = f32_to_bf16(in[i]);
}

// ---------------------------------------------------------------------------
// Router: logits = x @ gate^T, softmax, top-2, renormalize.
// One wave (32 lanes) per token.
// ---------------------------------------------------------------------------
__global__ void __launch_bounds__(128)
router_kernel(const float* __restrict__ x, const float* __restrict__ gate,
              int* __restrict__ sel, float* __restrict__ wgt) {
    int tok  = (blockIdx.x * 128 + threadIdx.x) >> 5;
    int lane = threadIdx.x & 31;
    if (tok >= NTOK) return;

    const float* xr = x + (size_t)tok * DD;
    float acc[EE];
#pragma unroll
    for (int e = 0; e < EE; ++e) acc[e] = 0.0f;

    for (int k = lane; k < DD; k += 32) {
        float xv = xr[k];
#pragma unroll
        for (int e = 0; e < EE; ++e) acc[e] += xv * gate[e * DD + k];
    }
#pragma unroll
    for (int e = 0; e < EE; ++e)
        for (int off = 16; off > 0; off >>= 1)
            acc[e] += __shfl_xor(acc[e], off, 32);

    if (lane == 0) {
        float mx = acc[0];
#pragma unroll
        for (int e = 1; e < EE; ++e) mx = fmaxf(mx, acc[e]);
        float p[EE], s = 0.0f;
#pragma unroll
        for (int e = 0; e < EE; ++e) { p[e] = __expf(acc[e] - mx); s += p[e]; }
        float inv = 1.0f / s;
#pragma unroll
        for (int e = 0; e < EE; ++e) p[e] *= inv;
        // top-2, ties -> lowest index (strict >)
        int i0 = 0;
#pragma unroll
        for (int e = 1; e < EE; ++e) if (p[e] > p[i0]) i0 = e;
        int i1 = (i0 == 0) ? 1 : 0;
#pragma unroll
        for (int e = 0; e < EE; ++e) if (e != i0 && p[e] > p[i1]) i1 = e;
        float sw = p[i0] + p[i1];
        sel[tok * 2 + 0] = i0;
        sel[tok * 2 + 1] = i1;
        wgt[tok * 2 + 0] = p[i0] / sw;
        wgt[tok * 2 + 1] = p[i1] / sw;
    }
}

// ---------------------------------------------------------------------------
// WMMA bf16 GEMM:  C[M,N] = A[M,K] @ W[N,K]^T (+ bias[N])
// A, W bf16 row-major (both contiguous in K). C f32 row-major.
// 1 wave -> 64x64 C tile (4x4 of 16x16 WMMA), 4 waves -> 128x128 block tile.
// M,N multiples of 128; K multiple of 32.
// ---------------------------------------------------------------------------
__global__ void __launch_bounds__(128)
wmma_gemm_bf16_kernel(const unsigned short* __restrict__ Abuf,
                      const unsigned short* __restrict__ Wbuf,
                      const float* __restrict__ bias,
                      float* __restrict__ Cbuf,
                      int M, int N, int K) {
    const int lane = threadIdx.x & 31;
    const int wave = threadIdx.x >> 5;
    const int m16  = lane & 15;
    const int half = lane >> 4;
    const int row0 = blockIdx.x * 128 + (wave >> 1) * 64;
    const int col0 = blockIdx.y * 128 + (wave & 1) * 64;
    (void)M;

    v8f acc[4][4];
#pragma unroll
    for (int mi = 0; mi < 4; ++mi)
#pragma unroll
        for (int ni = 0; ni < 4; ++ni) acc[mi][ni] = (v8f)(0.0f);

    for (int k0 = 0; k0 < K; k0 += 32) {
        v16bf afrag[4], bfrag[4];
#pragma unroll
        for (int mi = 0; mi < 4; ++mi) {
            // A lane layout: row = m16, K = half*16 + 0..15 (contiguous)
            const unsigned short* pa =
                Abuf + (size_t)(row0 + mi * 16 + m16) * K + k0 + half * 16;
            v8bf lo = *(const v8bf*)pa;
            v8bf hi = *(const v8bf*)(pa + 8);
            afrag[mi] = __builtin_shufflevector(lo, hi,
                0, 1, 2, 3, 4, 5, 6, 7, 8, 9, 10, 11, 12, 13, 14, 15);
        }
#pragma unroll
        for (int ni = 0; ni < 4; ++ni) {
            // B lane layout: col = m16, K = half*16 + 0..15 (contiguous)
            const unsigned short* pb =
                Wbuf + (size_t)(col0 + ni * 16 + m16) * K + k0 + half * 16;
            bfrag[ni] = *(const v16bf*)pb;
        }
#pragma unroll
        for (int mi = 0; mi < 4; ++mi)
#pragma unroll
            for (int ni = 0; ni < 4; ++ni)
                acc[mi][ni] = __builtin_amdgcn_wmma_f32_16x16x32_bf16(
                    false, afrag[mi], false, bfrag[ni],
                    (short)0, acc[mi][ni], false, false);
    }

#pragma unroll
    for (int mi = 0; mi < 4; ++mi)
#pragma unroll
        for (int ni = 0; ni < 4; ++ni) {
            int col = col0 + ni * 16 + m16;
            float b = bias ? bias[col] : 0.0f;
#pragma unroll
            for (int r = 0; r < 8; ++r) {
                int row = row0 + mi * 16 + half * 8 + r;   // C: VGPR r -> M = r + 8*half
                Cbuf[(size_t)row * N + col] = acc[mi][ni][r] + b;
            }
        }
}

// ---------------------------------------------------------------------------
// LoRA1 down: u1[t, r] = x[n] . A1[e][r]   (one wave per expanded row)
// ---------------------------------------------------------------------------
__global__ void __launch_bounds__(128)
lora1_down_kernel(const float* __restrict__ x, const float* __restrict__ A1,
                  const int* __restrict__ sel, float* __restrict__ u1) {
    int t    = (blockIdx.x * 128 + threadIdx.x) >> 5;
    int lane = threadIdx.x & 31;
    if (t >= TEXP) return;
    int n = t >> 1;
    int e = sel[t];
    const float* xr = x + (size_t)n * DD;
    const float* Ae = A1 + (size_t)e * RR * DD;

    float acc[RR];
#pragma unroll
    for (int r = 0; r < RR; ++r) acc[r] = 0.0f;
    for (int k = lane; k < DD; k += 32) {
        float xv = xr[k];
#pragma unroll
        for (int r = 0; r < RR; ++r) acc[r] += xv * Ae[r * DD + k];
    }
#pragma unroll
    for (int r = 0; r < RR; ++r)
        for (int off = 16; off > 0; off >>= 1)
            acc[r] += __shfl_xor(acc[r], off, 32);
    if (lane == 0) {
#pragma unroll
        for (int r = 0; r < RR; ++r) u1[(size_t)t * RR + r] = acc[r];
    }
}

// ---------------------------------------------------------------------------
// a[t,h] = bf16( silu( common_fc1[n,h] + scale * u1[t] . B1[e][h,:] ) )
// ---------------------------------------------------------------------------
__global__ void __launch_bounds__(256)
act_kernel(const float* __restrict__ fc1, const float* __restrict__ u1,
           const float* __restrict__ B1, const int* __restrict__ sel,
           unsigned short* __restrict__ abuf) {
    long idx = (long)blockIdx.x * 256 + threadIdx.x;
    if (idx >= (long)TEXP * HH) return;
    int t = (int)(idx / HH);
    int h = (int)(idx % HH);
    int n = t >> 1;
    int e = sel[t];
    const float* br = B1 + ((size_t)e * HH + h) * RR;
    const float* ur = u1 + (size_t)t * RR;
    float s = 0.0f;
#pragma unroll
    for (int r = 0; r < RR; ++r) s += ur[r] * br[r];
    float f = fc1[(size_t)n * HH + h] + LSCALE * s;
    float a = f / (1.0f + __expf(-f));           // silu
    abuf[idx] = f32_to_bf16(a);
}

// ---------------------------------------------------------------------------
// LoRA2 down: u2[t, r] = a[t] . A2[e][r]   (one wave per expanded row)
// ---------------------------------------------------------------------------
__global__ void __launch_bounds__(128)
lora2_down_kernel(const unsigned short* __restrict__ abuf,
                  const float* __restrict__ A2,
                  const int* __restrict__ sel, float* __restrict__ u2) {
    int t    = (blockIdx.x * 128 + threadIdx.x) >> 5;
    int lane = threadIdx.x & 31;
    if (t >= TEXP) return;
    int e = sel[t];
    const unsigned short* ar = abuf + (size_t)t * HH;
    const float* Ae = A2 + (size_t)e * RR * HH;

    float acc[RR];
#pragma unroll
    for (int r = 0; r < RR; ++r) acc[r] = 0.0f;
    for (int k = lane; k < HH; k += 32) {
        float av = bf16_to_f32(ar[k]);
#pragma unroll
        for (int r = 0; r < RR; ++r) acc[r] += av * Ae[r * HH + k];
    }
#pragma unroll
    for (int r = 0; r < RR; ++r)
        for (int off = 16; off > 0; off >>= 1)
            acc[r] += __shfl_xor(acc[r], off, 32);
    if (lane == 0) {
#pragma unroll
        for (int r = 0; r < RR; ++r) u2[(size_t)t * RR + r] = acc[r];
    }
}

// ---------------------------------------------------------------------------
// out[n,d] = sum_{k=0,1} w[t] * ( y[t,d] + b2[d] + scale * u2[t].B2[e][d,:] )
// ---------------------------------------------------------------------------
__global__ void __launch_bounds__(256)
final_kernel(const float* __restrict__ y, const float* __restrict__ u2,
             const float* __restrict__ B2, const float* __restrict__ b2,
             const int* __restrict__ sel, const float* __restrict__ wgt,
             float* __restrict__ out) {
    long idx = (long)blockIdx.x * 256 + threadIdx.x;
    if (idx >= (long)NTOK * DD) return;
    int n = (int)(idx / DD);
    int d = (int)(idx % DD);
    float res = 0.0f;
#pragma unroll
    for (int k = 0; k < 2; ++k) {
        int t = n * 2 + k;
        int e = sel[t];
        const float* br = B2 + ((size_t)e * DD + d) * RR;
        const float* ur = u2 + (size_t)t * RR;
        float s = 0.0f;
#pragma unroll
        for (int r = 0; r < RR; ++r) s += ur[r] * br[r];
        float yy = y[(size_t)t * DD + d] + b2[d] + LSCALE * s;
        res += wgt[t] * yy;
    }
    out[idx] = res;
}

// ---------------------------------------------------------------------------
// Host side
// ---------------------------------------------------------------------------
extern "C" void kernel_launch(void* const* d_in, const int* in_sizes, int n_in,
                              void* d_out, int out_size, void* d_ws, size_t ws_size,
                              hipStream_t stream) {
    (void)in_sizes; (void)n_in; (void)out_size; (void)ws_size;
    const float* x    = (const float*)d_in[0];   // [NTOK, DD]
    const float* gate = (const float*)d_in[1];   // [EE, DD]
    const float* W1   = (const float*)d_in[2];   // [HH, DD]
    const float* b1   = (const float*)d_in[3];   // [HH]
    const float* W2   = (const float*)d_in[4];   // [DD, HH]
    const float* b2   = (const float*)d_in[5];   // [DD]
    const float* A1   = (const float*)d_in[6];   // [EE, RR, DD]
    const float* B1   = (const float*)d_in[7];   // [EE, HH, RR]
    const float* A2   = (const float*)d_in[8];   // [EE, RR, HH]
    const float* B2   = (const float*)d_in[9];   // [EE, DD, RR]
    float* out = (float*)d_out;

    // Workspace layout (all offsets keep >= 256B alignment)
    char* ws = (char*)d_ws;
    size_t off = 0;
    unsigned short* x_bf  = (unsigned short*)(ws + off); off += (size_t)NTOK * DD * 2;  // 4 MB
    unsigned short* w1_bf = (unsigned short*)(ws + off); off += (size_t)HH * DD * 2;    // 8 MB
    unsigned short* w2_bf = (unsigned short*)(ws + off); off += (size_t)DD * HH * 2;    // 8 MB
    float*          fc1   = (float*)(ws + off);          off += (size_t)NTOK * HH * 4;  // 32 MB
    unsigned short* abuf  = (unsigned short*)(ws + off); off += (size_t)TEXP * HH * 2;  // 32 MB
    float*          ybuf  = (float*)(ws + off);          off += (size_t)TEXP * DD * 4;  // 16 MB
    float*          u1    = (float*)(ws + off);          off += (size_t)TEXP * RR * 4;
    float*          u2    = (float*)(ws + off);          off += (size_t)TEXP * RR * 4;
    int*            sel   = (int*)(ws + off);            off += (size_t)TEXP * 4;
    float*          wgt   = (float*)(ws + off);          off += (size_t)TEXP * 4;

    // 1) bf16 conversions
    cvt_f32_bf16_kernel<<<(NTOK * DD + 255) / 256, 256, 0, stream>>>(x, x_bf, NTOK * DD);
    cvt_f32_bf16_kernel<<<(HH * DD + 255) / 256, 256, 0, stream>>>(W1, w1_bf, HH * DD);
    cvt_f32_bf16_kernel<<<(DD * HH + 255) / 256, 256, 0, stream>>>(W2, w2_bf, DD * HH);

    // 2) router (one wave per token)
    router_kernel<<<NTOK / 4, 128, 0, stream>>>(x, gate, sel, wgt);

    // 3) common_fc1 = x @ W1^T + b1   [2048 x 4096, K=1024]
    wmma_gemm_bf16_kernel<<<dim3(NTOK / 128, HH / 128), 128, 0, stream>>>(
        x_bf, w1_bf, b1, fc1, NTOK, HH, DD);

    // 4) LoRA1 down
    lora1_down_kernel<<<TEXP / 4, 128, 0, stream>>>(x, A1, sel, u1);

    // 5) fused LoRA1 up + SiLU -> bf16 activations
    act_kernel<<<(int)(((long)TEXP * HH + 255) / 256), 256, 0, stream>>>(
        fc1, u1, B1, sel, abuf);

    // 6) y = a @ W2^T   [4096 x 1024, K=4096]  (b2 added in final)
    wmma_gemm_bf16_kernel<<<dim3(TEXP / 128, DD / 128), 128, 0, stream>>>(
        abuf, w2_bf, nullptr, ybuf, TEXP, DD, HH);

    // 7) LoRA2 down
    lora2_down_kernel<<<TEXP / 4, 128, 0, stream>>>(abuf, A2, sel, u2);

    // 8) fused LoRA2 up + bias + top-2 weighted reduce
    final_kernel<<<(int)(((long)NTOK * DD + 255) / 256), 256, 0, stream>>>(
        ybuf, u2, B2, b2, sel, wgt, out);
}
